// CustomConvModule_29085518529124
// MI455X (gfx1250) — matrix operand order repaired and checked by reading, not compiled, not run on hardware
//
#include <hip/hip_runtime.h>

// ---------------------------------------------------------------------------
// Implicit-GEMM 3x3 conv for MI455X (gfx1250, wave32, WMMA).
//   M = B*H*W output pixels, N = 256 out channels, K = 128*9 = 1152
//   bf16 A/B fragments, f32 accumulation via v_wmma_f32_16x16x32_bf16.
//   One workgroup = one (b, h) output row: 56 px (4 strips of 16) x 256 chans.
//   Halo rows (h-1,h,h+1) x 66 px x 128 ch staged to LDS as bf16 ONCE; the 9
//   unfold patch views are shifted LDS slices -> barrier-free K loop.
//   Each wave: 4 M-strips x 2 N-tiles = 8 accumulators; B reused 4x per load.
// ---------------------------------------------------------------------------

typedef __attribute__((ext_vector_type(16))) __bf16 v16bf;
typedef __attribute__((ext_vector_type(8)))  __bf16 v8bf;
typedef __attribute__((ext_vector_type(8)))  float  v8f;

static constexpr int Bn   = 32;    // batch
static constexpr int Cin  = 128;   // in channels
static constexpr int Hh   = 56;
static constexpr int Ww   = 56;
static constexpr int Oc   = 256;   // out channels
static constexpr int Ktot = Cin * 9;       // 1152
static constexpr int KSTEPS = Ktot / 32;   // 36 wmma k-steps
static constexpr int PXE  = 66;    // extended pixels: w = -1 .. 64
static constexpr int PSTR = 136;   // LDS pixel stride in halfs (272B, bank-safe)

__device__ __forceinline__ unsigned bf16rne(float f) {
  unsigned u = __builtin_bit_cast(unsigned, f);
  return u + 0x7FFFu + ((u >> 16) & 1u);   // rounded bits live in [31:16]
}

// ---------------------------------------------------------------------------
// Pack f32 weights (O, C, 3, 3) into bf16 B-fragments in workspace,
// fragment-major: ws[ntile][kstep][lane][elem]  (16 * 36 * 32 * 16 bf16).
// B-matrix (32x16 bf16) per-lane layout: N = lane%16, half = lane/16,
//   element e -> K = kstep*32 + 16*half + e,  with K = p*128 + c.
// ---------------------------------------------------------------------------
__global__ __launch_bounds__(256) void pack_weights_bf16(
    const float* __restrict__ w, unsigned short* __restrict__ wsu) {
  int idx = blockIdx.x * 256 + threadIdx.x;
  constexpr int total = (Oc / 16) * KSTEPS * 32 * 16;
  if (idx >= total) return;
  int e    = idx & 15;
  int lane = (idx >> 4) & 31;
  int sn   = idx >> 9;
  int s    = sn % KSTEPS;
  int n    = sn / KSTEPS;
  int o = n * 16 + (lane & 15);
  int K = s * 32 + (lane >> 4) * 16 + e;
  int c = K & 127;       // K = p*128 + c
  int p = K >> 7;        // patch position 0..8 (di*3+dj)
  float v = w[(o * Cin + c) * 9 + p];
  wsu[idx] = (unsigned short)(bf16rne(v) >> 16);
}

// ---------------------------------------------------------------------------
__global__ __launch_bounds__(256) void conv3x3_wmma_bf16(
    const float* __restrict__ x,
    const unsigned short* __restrict__ wsu,
    const float* __restrict__ bias,
    float* __restrict__ out) {
  const __bf16* __restrict__ wpack = reinterpret_cast<const __bf16*>(wsu);

  // 3 halo rows x 66 pixels x 128 channels, bf16 (+pad) = 53.9 KB
  __shared__ __bf16 As[3 * PXE * PSTR];

  const int tid  = threadIdx.x;
  const int wave = tid >> 5;
  const int lane = tid & 31;
  const int lm   = lane & 15;   // A row / C column index
  const int lh   = lane >> 4;   // half-wave select

  // one block per (b, h) output row
  const int hrow = blockIdx.x % Hh;
  const int bidx = blockIdx.x / Hh;

  const int n0 = wave * 2;      // this wave's two N tiles: n0, n0+1

  v8f acc[4][2];
#pragma unroll
  for (int st = 0; st < 4; ++st) {
    acc[st][0] = v8f{};
    acc[st][1] = v8f{};
  }

  // ---- stage halo rows once: branch-free clamp + 0/1 mask, pairs of chans.
  // flat element L = ((row*64 + chanpair)*66 + pxi): consecutive threads ->
  // consecutive pixels -> coalesced f32 loads.
  constexpr int stageTotal = 3 * 64 * PXE;         // 12672 dword stores
  constexpr int stageIters = (stageTotal + 255) / 256;  // 50
  for (int j = 0; j < stageIters; ++j) {
    const int L = tid + 256 * j;
    if (L < stageTotal) {
      const int pxi = L % PXE;
      const int rem = L / PXE;
      const int cp  = rem & 63;
      const int row = rem >> 6;          // 0..2 -> di
      const int hin = hrow + row - 1;
      const int wv  = pxi - 1;
      const bool valid = (hin >= 0) && (hin < Hh) && (wv >= 0) && (wv < Ww);
      const int hc = min(max(hin, 0), Hh - 1);
      const int wc = min(max(wv, 0), Ww - 1);
      const float msk = valid ? 1.0f : 0.0f;
      const int cc = cp * 2;
      const float* pb = x + (((long)bidx * Cin + cc) * Hh + hc) * Ww + wc;
      const float v0 = pb[0] * msk;
      const float v1 = pb[Hh * Ww] * msk;          // channel cc+1
      const unsigned packed =
          (bf16rne(v0) >> 16) | (bf16rne(v1) & 0xFFFF0000u);
      *reinterpret_cast<unsigned*>(&As[(row * PXE + pxi) * PSTR + cc]) = packed;
    }
  }
  __syncthreads();   // the only barrier: K loop below is barrier-free

  // ---- K loop: 9 patch positions x 4 k-substeps, 8 WMMAs each ----
  for (int p = 0; p < 9; ++p) {
    const int di = p / 3;      // halo row 0..2
    const int dj = p % 3;      // pixel shift 0..2 (w offset dj-1, pxi offset dj)
#pragma unroll
    for (int sub = 0; sub < 4; ++sub) {
      const int k0 = sub * 32;
      const int s  = p * 4 + sub;

      // B fragments: contiguous 32B per lane, pre-packed (L2 hot)
      const v16bf b0 = *reinterpret_cast<const v16bf*>(
          wpack + (((n0 + 0) * KSTEPS + s) * 32 + lane) * 16);
      const v16bf b1 = *reinterpret_cast<const v16bf*>(
          wpack + (((n0 + 1) * KSTEPS + s) * 32 + lane) * 16);

#pragma unroll
      for (int st = 0; st < 4; ++st) {
        // A fragment, contiguous k order, as in B pack: K(e,half) = e + 16*half
        const int pxi  = st * 16 + lm + dj;
        const int abase = (di * PXE + pxi) * PSTR + k0 + 16 * lh;
        v8bf alo = *reinterpret_cast<const v8bf*>(&As[abase]);
        v8bf ahi = *reinterpret_cast<const v8bf*>(&As[abase + 8]);
        v16bf a = __builtin_shufflevector(alo, ahi, 0, 1, 2, 3, 4, 5, 6, 7,
                                                    8, 9, 10, 11, 12, 13, 14, 15);
        acc[st][0] = __builtin_amdgcn_wmma_f32_16x16x32_bf16(
            false, a, false, b0, (short)0, acc[st][0], false, false);
        acc[st][1] = __builtin_amdgcn_wmma_f32_16x16x32_bf16(
            false, a, false, b1, (short)0, acc[st][1], false, false);
      }
    }
  }

  // C/D layout: element r, lane -> M = r + 8*lh (pixel), N = lm (channel)
  const int o0 = n0 * 16 + lm;
  const int o1 = o0 + 16;
  const float bv0 = bias[o0];
  const float bv1 = bias[o1];
  float* outr0 = out + (((long)bidx * Oc + o0) * Hh + hrow) * Ww;
  float* outr1 = out + (((long)bidx * Oc + o1) * Hh + hrow) * Ww;

#pragma unroll
  for (int st = 0; st < 4; ++st) {
#pragma unroll
    for (int r = 0; r < 8; ++r) {
      const int wcol = st * 16 + r + 8 * lh;
      if (wcol < Ww) {
        outr0[wcol] = acc[st][0][r] + bv0;
        outr1[wcol] = acc[st][1][r] + bv1;
      }
    }
  }
}

// ---------------------------------------------------------------------------
extern "C" void kernel_launch(void* const* d_in, const int* in_sizes, int n_in,
                              void* d_out, int out_size, void* d_ws, size_t ws_size,
                              hipStream_t stream) {
  (void)in_sizes; (void)n_in; (void)out_size; (void)ws_size;
  const float* x      = (const float*)d_in[0];  // (32,128,56,56)
  const float* weight = (const float*)d_in[1];  // (256,128,3,3)
  const float* bias   = (const float*)d_in[2];  // (256,)
  float* out          = (float*)d_out;          // (32,256,56,56)
  unsigned short* wsu = (unsigned short*)d_ws;  // bf16 packed weights (576 KB)

  constexpr int packTotal = (Oc / 16) * KSTEPS * 32 * 16;   // 294912
  pack_weights_bf16<<<(packTotal + 255) / 256, 256, 0, stream>>>(weight, wsu);

  const int rows = Bn * Hh;                                 // 1792 workgroups
  conv3x3_wmma_bf16<<<rows, 256, 0, stream>>>(x, wsu, bias, out);
}
